// DeformConv_39651138077111
// MI455X (gfx1250) — matrix operand order repaired and checked by reading, not compiled, not run on hardware
//
#include <hip/hip_runtime.h>
#include <hip/hip_bf16.h>
#include <math.h>

typedef __attribute__((ext_vector_type(16))) _Float16 v16h;
typedef __attribute__((ext_vector_type(8)))  _Float16 v8h;
typedef __attribute__((ext_vector_type(8)))  float    v8f;
typedef int v4i __attribute__((vector_size(16)));

#define CC   256      // input channels
#define OO   256      // output channels
#define HH   64
#define WW   64
#define KT   9        // 3x3 taps
#define KDIM (KT*CC)  // 2304 GEMM K (k-major: k*256+c)

#if defined(__HIP_DEVICE_COMPILE__) && \
    __has_builtin(__builtin_amdgcn_global_load_async_to_lds_b128) && \
    __has_builtin(__builtin_amdgcn_s_wait_asynccnt)
#define USE_ASYNC_LDS 1
#else
#define USE_ASYNC_LDS 0
#endif

#define AS_GLOBAL __attribute__((address_space(1)))
#define AS_LDS    __attribute__((address_space(3)))

// ---------------------------------------------------------------------------
// Phase 0: offset conv (27 ch, 3x3) -> dy, dx, sigmoid(mask) into workspace
// One block per (b,h) row. LDS stages 3 input rows per channel.
// ---------------------------------------------------------------------------
__global__ __launch_bounds__(256) void offset_conv_kernel(
    const float* __restrict__ x, const float* __restrict__ w_off,
    const float* __restrict__ b_off,
    float* __restrict__ dyA, float* __restrict__ dxA, float* __restrict__ mkA)
{
    __shared__ float xs[3 * 66];   // 3 rows, 1 zero-pad col each side
    const int blk = blockIdx.x;    // b*64 + h
    const int b = blk >> 6;
    const int h = blk & 63;
    const int t = threadIdx.x;
    const int pix = t & 63;
    const int g = t >> 6;          // 0..3 (wave-pair uniform)

    if (t < 3) { xs[t * 66] = 0.f; xs[t * 66 + 65] = 0.f; }

    float acc[7];
#pragma unroll
    for (int j = 0; j < 7; j++) acc[j] = 0.f;

    const float* xb = x + (size_t)b * CC * HH * WW;
    for (int c = 0; c < CC; c++) {
        __syncthreads();
        if (t < 192) {
            int r = t / 64, w = t % 64;
            int hh = h - 1 + r;
            float v = 0.f;
            if (hh >= 0 && hh < HH) v = xb[(size_t)c * HH * WW + hh * WW + w];
            xs[r * 66 + 1 + w] = v;
        }
        __syncthreads();
        float xv[9];
#pragma unroll
        for (int ky = 0; ky < 3; ky++)
#pragma unroll
            for (int kx = 0; kx < 3; kx++)
                xv[ky * 3 + kx] = xs[ky * 66 + pix + kx];
#pragma unroll
        for (int j = 0; j < 7; j++) {
            int ch = g * 7 + j;
            if (ch < 27) {
                const float* wp = w_off + (size_t)ch * CC * 9 + c * 9;
#pragma unroll
                for (int tq = 0; tq < 9; tq++)
                    acc[j] = fmaf(wp[tq], xv[tq], acc[j]);
            }
        }
    }
#pragma unroll
    for (int j = 0; j < 7; j++) {
        int ch = g * 7 + j;
        if (ch < 27) {
            float om = acc[j] + b_off[ch];
            int k9 = ch % 9;
            size_t oidx = (((size_t)b * KT + k9) * HH + h) * WW + pix;
            if (ch < 9)       dyA[oidx] = om;
            else if (ch < 18) dxA[oidx] = om;
            else              mkA[oidx] = 1.f / (1.f + __expf(-om));
        }
    }
}

// ---------------------------------------------------------------------------
// Phase 1: weight f32(O,C,3,3) -> f16 with K-dim transposed to k-major:
//          wf[o*2304 + k*256 + c] = weight[o*2304 + c*9 + k]
// ---------------------------------------------------------------------------
__global__ __launch_bounds__(256) void wcvt_kernel(
    const float* __restrict__ w, _Float16* __restrict__ wf, int n)
{
    int i = blockIdx.x * 256 + threadIdx.x;
    if (i >= n) return;
    int o = i / KDIM;
    int r = i - o * KDIM;
    int k = r >> 8;      // /256
    int c = r & 255;
    wf[i] = (_Float16)w[(size_t)o * KDIM + c * 9 + k];
}

// ---------------------------------------------------------------------------
// Phase 2: fused deformable-sample + WMMA GEMM + bias/BN/ReLU.
// Block tile: 128 out-channels x 64 pixels (one image row), 8 waves (4x2),
// each wave owns a 32x32 tile = 4 x v_wmma_f32_16x16x32_f16 accumulators.
// A-slab is copied global->LDS via the async-LDS path (ASYNCcnt) when the
// toolchain exposes it, overlapping with the bilinear sampling VALU work.
// ---------------------------------------------------------------------------
__global__ __launch_bounds__(256) void deform_wmma_kernel(
    const float* __restrict__ x,
    const float* __restrict__ dyA, const float* __restrict__ dxA,
    const float* __restrict__ mkA, const _Float16* __restrict__ wf,
    const float* __restrict__ bias, const float* __restrict__ gamma,
    const float* __restrict__ beta, const float* __restrict__ rmean,
    const float* __restrict__ rvar, float* __restrict__ out)
{
    __shared__ short sy0[576], sy1[576], sx0[576], sx1[576];
    __shared__ float sw00[576], sw01[576], sw10[576], sw11[576];
    __shared__ _Float16 Als[128 * 32];   // A slab: 128 rows x 32 K (f16)
    __shared__ _Float16 Bls[64 * 40];    // B slab: 64 cols x 32 K, pad to 40

    const int blk  = blockIdx.x;         // (b*64+h)*2 + mblk
    const int mblk = blk & 1;
    const int bh   = blk >> 1;
    const int b    = bh >> 6;
    const int h    = bh & 63;
    const int t    = threadIdx.x;
    const int lane = t & 31;
    const int wave = t >> 5;
    const int mw   = wave & 3;           // 0..3 -> 32-row band
    const int nw   = wave >> 2;          // 0..1 -> 32-col band
    const int m0   = mblk * 128;

    // ---- stage bilinear params (per tap k, per pixel p; independent of c)
    for (int i = t; i < 576; i += 256) {
        int k = i / 64;
        int p = i - k * 64;
        int ky = k / 3, kx = k - ky * 3;
        size_t oidx = (((size_t)b * KT + k) * HH + h) * WW + p;
        float dyv = dyA[oidx], dxv = dxA[oidx], m = mkA[oidx];
        float py = dyv + (float)(h - 1 + ky);
        float px = dxv + (float)(p - 1 + kx);
        float y0f = floorf(py), x0f = floorf(px);
        float ly = py - y0f, lx = px - x0f;
        int y0 = (int)y0f, x0 = (int)x0f;
        int y1 = y0 + 1, x1 = x0 + 1;
        float vy0 = (y0 >= 0 && y0 < HH) ? 1.f : 0.f;
        float vy1 = (y1 >= 0 && y1 < HH) ? 1.f : 0.f;
        float vx0 = (x0 >= 0 && x0 < WW) ? 1.f : 0.f;
        float vx1 = (x1 >= 0 && x1 < WW) ? 1.f : 0.f;
        sw00[i] = (1.f - ly) * (1.f - lx) * m * vy0 * vx0;
        sw01[i] = (1.f - ly) * lx * m * vy0 * vx1;
        sw10[i] = ly * (1.f - lx) * m * vy1 * vx0;
        sw11[i] = ly * lx * m * vy1 * vx1;
        sy0[i] = (short)min(max(y0, 0), HH - 1);
        sy1[i] = (short)min(max(y1, 0), HH - 1);
        sx0[i] = (short)min(max(x0, 0), WW - 1);
        sx1[i] = (short)min(max(x1, 0), WW - 1);
    }
    __syncthreads();

    v8f acc00 = {}, acc01 = {}, acc10 = {}, acc11 = {};

    const int p  = t & 63;   // pixel this thread generates B-values for
    const int cg = t >> 6;   // channel group 0..3 (8 channels each)
    const float* xb = x + (size_t)b * CC * HH * WW;

    for (int k = 0; k < KT; k++) {
        const int pi = k * 64 + p;
        const int y0 = sy0[pi], y1 = sy1[pi], x0 = sx0[pi], x1 = sx1[pi];
        const float w00 = sw00[pi], w01 = sw01[pi];
        const float w10 = sw10[pi], w11 = sw11[pi];
        const int o00 = y0 * WW + x0, o01 = y0 * WW + x1;
        const int o10 = y1 * WW + x0, o11 = y1 * WW + x1;

        for (int cb = 0; cb < 8; cb++) {        // 32-channel K-slabs
            __syncthreads();                     // LDS reuse fence
            // ---- copy A slab: 128 rows x 64B from f16 weight table
            {
                const _Float16* asrc = wf + (size_t)k * 256 + cb * 32;
#pragma unroll
                for (int q = 0; q < 2; q++) {
                    int ci = t + q * 256;        // 512 x 16B chunks
                    int row = ci >> 2;
                    int part = ci & 3;
                    const _Float16* src =
                        asrc + (size_t)(m0 + row) * KDIM + part * 8;
                    _Float16* dst = &Als[row * 32 + part * 8];
#if USE_ASYNC_LDS
                    __builtin_amdgcn_global_load_async_to_lds_b128(
                        (AS_GLOBAL v4i*)(uintptr_t)src,
                        (AS_LDS v4i*)(uint32_t)(uintptr_t)dst, 0, 0);
#else
                    *(uint4*)dst = *(const uint4*)src;
#endif
                }
                if (cb < 7) {   // prefetch next A slab (global_prefetch_b8)
                    const _Float16* nsrc =
                        wf + (size_t)k * 256 + (cb + 1) * 32 +
                        (size_t)(m0 + (t >> 1)) * KDIM;
                    __builtin_prefetch(nsrc, 0, 2);
                }
            }
            // ---- generate deformed-sample B slab (f16), 8 values/thread
#pragma unroll
            for (int cc = 0; cc < 8; cc++) {
                int c = cb * 32 + cg * 8 + cc;
                const float* pl = xb + (size_t)c * (HH * WW);
                float v = w00 * pl[o00] + w01 * pl[o01]
                        + w10 * pl[o10] + w11 * pl[o11];
                Bls[p * 40 + cg * 8 + cc] = (_Float16)v;
            }
#if USE_ASYNC_LDS
            __builtin_amdgcn_s_wait_asynccnt(0);
#endif
            __syncthreads();

            // ---- assemble fragments per ISA 7.12.2 layouts and WMMA
            const int base0 = (lane < 16) ? 0 : 8;       // A K-base
            const int arow0 = mw * 32 + (lane & 15);
            const int arow1 = arow0 + 16;
            v8h alo0 = *(const v8h*)(&Als[arow0 * 32 + base0]);
            v8h ahi0 = *(const v8h*)(&Als[arow0 * 32 + base0 + 16]);
            v8h alo1 = *(const v8h*)(&Als[arow1 * 32 + base0]);
            v8h ahi1 = *(const v8h*)(&Als[arow1 * 32 + base0 + 16]);
            v16h a0 = __builtin_shufflevector(alo0, ahi0,
                0,1,2,3,4,5,6,7,8,9,10,11,12,13,14,15);
            v16h a1 = __builtin_shufflevector(alo1, ahi1,
                0,1,2,3,4,5,6,7,8,9,10,11,12,13,14,15);

            const int kb    = (lane < 16) ? 0 : 8;       // B K-base
            const int bcol0 = nw * 32 + (lane & 15);
            const int bcol1 = bcol0 + 16;
            v8h blo0 = *(const v8h*)(&Bls[bcol0 * 40 + kb]);
            v8h bhi0 = *(const v8h*)(&Bls[bcol0 * 40 + kb + 16]);
            v8h blo1 = *(const v8h*)(&Bls[bcol1 * 40 + kb]);
            v8h bhi1 = *(const v8h*)(&Bls[bcol1 * 40 + kb + 16]);
            v16h b0 = __builtin_shufflevector(blo0, bhi0,
                0,1,2,3,4,5,6,7,8,9,10,11,12,13,14,15);
            v16h b1 = __builtin_shufflevector(blo1, bhi1,
                0,1,2,3,4,5,6,7,8,9,10,11,12,13,14,15);

            acc00 = __builtin_amdgcn_wmma_f32_16x16x32_f16(
                false, a0, false, b0, (short)0, acc00, false, false);
            acc01 = __builtin_amdgcn_wmma_f32_16x16x32_f16(
                false, a0, false, b1, (short)0, acc01, false, false);
            acc10 = __builtin_amdgcn_wmma_f32_16x16x32_f16(
                false, a1, false, b0, (short)0, acc10, false, false);
            acc11 = __builtin_amdgcn_wmma_f32_16x16x32_f16(
                false, a1, false, b1, (short)0, acc11, false, false);
        }
    }

    // ---- epilogue: bias + BN(affine, inference) + ReLU, direct store
    const int l15 = lane & 15;
    const int hi8 = (lane >> 4) * 8;    // C/D layout: VGPR r -> M=r or r+8
#pragma unroll
    for (int mt = 0; mt < 2; mt++) {
#pragma unroll
        for (int r = 0; r < 8; r++) {
            int o = m0 + mw * 32 + mt * 16 + r + hi8;
            float sc = gamma[o] * rsqrtf(rvar[o] + 1e-5f);
            float sh = sc * (bias[o] - rmean[o]) + beta[o];
            float va = (mt == 0) ? acc00[r] : acc10[r];
            float vb = (mt == 0) ? acc01[r] : acc11[r];
            float r0 = fmaxf(fmaf(sc, va, sh), 0.f);
            float r1 = fmaxf(fmaf(sc, vb, sh), 0.f);
            size_t ob = (((size_t)b * OO + o) * HH + h) * WW;
            out[ob + nw * 32 + l15]      = r0;
            out[ob + nw * 32 + 16 + l15] = r1;
        }
    }
}

// ---------------------------------------------------------------------------
extern "C" void kernel_launch(void* const* d_in, const int* in_sizes, int n_in,
                              void* d_out, int out_size, void* d_ws, size_t ws_size,
                              hipStream_t stream)
{
    const float* x      = (const float*)d_in[0];
    const float* w_off  = (const float*)d_in[1];
    const float* b_off  = (const float*)d_in[2];
    const float* weight = (const float*)d_in[3];
    const float* bias   = (const float*)d_in[4];
    const float* gamma  = (const float*)d_in[5];
    const float* beta   = (const float*)d_in[6];
    const float* rmean  = (const float*)d_in[7];
    const float* rvar   = (const float*)d_in[8];
    float* outp = (float*)d_out;

    const int BHW9 = 4 * KT * HH * WW;          // 147456 floats per field
    float* dyA = (float*)d_ws;
    float* dxA = dyA + BHW9;
    float* mkA = dxA + BHW9;
    _Float16* wf = (_Float16*)(mkA + BHW9);     // 589824 halves

    // Phase 0: offset conv -> dy/dx/mask          (B*H = 256 blocks)
    offset_conv_kernel<<<4 * HH, 256, 0, stream>>>(x, w_off, b_off, dyA, dxA, mkA);
    // Phase 1: weight f32 -> f16, k-major K-dim   (589824 elems)
    wcvt_kernel<<<(OO * KDIM + 255) / 256, 256, 0, stream>>>(weight, wf, OO * KDIM);
    // Phase 2: fused sample + WMMA GEMM + BN/ReLU (B*H*2 = 512 blocks)
    deform_wmma_kernel<<<4 * HH * 2, 256, 0, stream>>>(
        x, dyA, dxA, mkA, wf, bias, gamma, beta, rmean, rvar, outp);
}
